// CompressiveMemory_88888643158561
// MI455X (gfx1250) — matrix operand order repaired and checked by reading, not compiled, not run on hardware
//
#include <hip/hip_runtime.h>

#define S_TOT 65536
#define D_IN  128
#define EKD   256
#define EVD   256

typedef __attribute__((ext_vector_type(16))) __bf16          v16bf;
typedef __attribute__((ext_vector_type(8)))  __bf16          v8bf;
typedef __attribute__((ext_vector_type(8)))  float           v8f;
typedef __attribute__((ext_vector_type(8)))  unsigned short  us8;

static __device__ inline unsigned short f2bf(float f) {
  unsigned int u = __float_as_uint(f);
  u += 0x7fffu + ((u >> 16) & 1u);          // round-to-nearest-even
  return (unsigned short)(u >> 16);
}

static __device__ inline float wave_sum(float v) {
#pragma unroll
  for (int m = 16; m >= 1; m >>= 1) v += __shfl_xor(v, m, 32);
  return v;
}

static __device__ inline v16bf cat8(v8bf lo, v8bf hi) {
  return __builtin_shufflevector(lo, hi, 0,1,2,3,4,5,6,7,8,9,10,11,12,13,14,15);
}

// ---------------------------------------------------------------------------
// Kernel: bf16 transpose of memory [EK][EV] -> Bt[n][k] (n-major, k contiguous)
// ---------------------------------------------------------------------------
__global__ __launch_bounds__(256) void prep_bt_kernel(
    const float* __restrict__ mem, unsigned short* __restrict__ Bt) {
  int n = blockIdx.x;            // EV column
  int t = threadIdx.x;           // EK row
  Bt[n * EKD + t] = f2bf(mem[(size_t)t * EVD + n]);
}

// ---------------------------------------------------------------------------
// Kernel: per-row dpfp activation + sum-normalize (exact f32).
//   A_bf[s][i]  = bf16(act(x)[s][i])
//   invden[s]   = 1 / (act(x)[s] . zn + 1e-8)
//   zn_out[s][i]= zn[i] + act(x)[s][i]        (only when zn_out != nullptr)
// One wave per row; 8 rows per 256-thread block.
// ---------------------------------------------------------------------------
__global__ __launch_bounds__(256) void act_kernel(
    const float* __restrict__ x, const float* __restrict__ zn,
    unsigned short* __restrict__ A_bf, float* __restrict__ invden,
    float* __restrict__ zn_out) {
  const int wave = threadIdx.x >> 5;
  const int lane = threadIdx.x & 31;
  const int s = blockIdx.x * 8 + wave;
  const float* xr = x + (size_t)s * D_IN;
  const int i0 = lane * 8;

  // x(j) = j<128 ? relu(q[j]) : relu(-q[j-128]),  f(i) = x(i)*x(i-1 mod 256)
  float f[8];
  float sum = 0.f;
  {
    int jp = (i0 + 255) & 255;
    float vq = xr[jp & 127];
    float xprev = (jp < D_IN) ? fmaxf(vq, 0.f) : fmaxf(-vq, 0.f);
#pragma unroll
    for (int t = 0; t < 8; ++t) {
      int j = i0 + t;
      float vv = xr[j & 127];
      float xc = (j < D_IN) ? fmaxf(vv, 0.f) : fmaxf(-vv, 0.f);
      f[t] = xc * xprev;
      xprev = xc;
      sum += f[t];
    }
  }
  sum = wave_sum(sum);
  const float inv = 1.0f / sum;

  float znv[8], av[8];
  float den = 0.f;
#pragma unroll
  for (int t = 0; t < 8; ++t) {
    znv[t] = zn[i0 + t];
    av[t]  = f[t] * inv;
    den   += av[t] * znv[t];
  }
  den = wave_sum(den) + 1e-8f;
  if (lane == 0) invden[s] = 1.0f / den;

  us8 pack;
#pragma unroll
  for (int t = 0; t < 8; ++t) pack[t] = f2bf(av[t]);
  *(us8*)(A_bf + (size_t)s * EKD + i0) = pack;

  if (zn_out) {
    float4 o0 = make_float4(znv[0] + av[0], znv[1] + av[1], znv[2] + av[2], znv[3] + av[3]);
    float4 o1 = make_float4(znv[4] + av[4], znv[5] + av[5], znv[6] + av[6], znv[7] + av[7]);
    *(float4*)(zn_out + (size_t)s * EKD + i0)     = o0;
    *(float4*)(zn_out + (size_t)s * EKD + i0 + 4) = o1;
  }
}

// ---------------------------------------------------------------------------
// Kernel: retrieve = (A @ memory) * invden   [per-row scale]
//   mode a_mem (vsrc==nullptr): out_f32[s][n] = num*invden[s]
//   mode W     (vsrc!=nullptr): out_bf[s][n]  = bf16(v[s][n] - num*invden[s])
// Block = 256 thr (8 waves), computes a 64-row x 256-col tile.
// Pure-register WMMA GEMM: A fragments from global bf16 rows, B fragments from
// the 128 KB transposed-bf16 memory copy (L2/WGP$-resident on 192 MB L2).
// ---------------------------------------------------------------------------
__global__ __launch_bounds__(256) void retrieve_kernel(
    const unsigned short* __restrict__ A_bf, const float* __restrict__ invden,
    const unsigned short* __restrict__ Bt,
    float* __restrict__ out_f32, unsigned short* __restrict__ out_bf,
    const float* __restrict__ vsrc) {
  const int wave = threadIdx.x >> 5;
  const int lane = threadIdx.x & 31;
  const int rt = wave & 3;        // row tile 0..3
  const int ch = wave >> 2;       // column half 0..1
  const int h  = lane >> 4;       // lane half
  const int ln = lane & 15;

  const int row_base = blockIdx.x * 64 + rt * 16;   // 16 output rows
  const unsigned short* Arow = A_bf + (size_t)(row_base + ln) * EKD;

  v8f acc[8];
#pragma unroll
  for (int c = 0; c < 8; ++c) acc[c] = (v8f){0.f,0.f,0.f,0.f,0.f,0.f,0.f,0.f};

#pragma unroll 2
  for (int kk = 0; kk < EKD; kk += 32) {
    __builtin_prefetch(Arow + kk + 32, 0, 3);       // WGP-scope prefetch
    const v8bf alo = *(const v8bf*)(Arow + kk + 8 * h);
    const v8bf ahi = *(const v8bf*)(Arow + kk + 16 + 8 * h);
    const v16bf a = cat8(alo, ahi);
#pragma unroll
    for (int ct = 0; ct < 8; ++ct) {
      const int n = ch * 128 + ct * 16 + ln;
      const unsigned short* bp = Bt + (size_t)n * EKD + kk + 8 * h;
      const v16bf b = cat8(*(const v8bf*)bp, *(const v8bf*)(bp + 16));
      acc[ct] = __builtin_amdgcn_wmma_f32_16x16x32_bf16(
          false, a, false, b, (short)0, acc[ct], false, false);
    }
  }

  float idv[8];
#pragma unroll
  for (int r = 0; r < 8; ++r) idv[r] = invden[row_base + 8 * h + r];

#pragma unroll
  for (int ct = 0; ct < 8; ++ct) {
    const int n = ch * 128 + ct * 16 + ln;
#pragma unroll
    for (int r = 0; r < 8; ++r) {
      const int srow = row_base + 8 * h + r;       // C layout: M = r + 8h
      float val = acc[ct][r] * idv[r];
      if (vsrc) {
        val = vsrc[(size_t)srow * EVD + n] - val;
        out_bf[(size_t)srow * EVD + n] = f2bf(val);
      } else {
        out_f32[(size_t)srow * EVD + n] = val;
      }
    }
  }
}

// ---------------------------------------------------------------------------
// Kernel: partial to_add = Ak[chunk]^T @ W[chunk]   -> partials[block][EK][EV]
// Block = 256 thr (8 waves) owns a full 256x256 f32 accumulator.
// 32-row S-chunks are staged through LDS with on-the-fly transpose so both
// WMMA operands get contiguous-K fragments (ds_load_b128).
// Software-pipelined: chunk i+1's global_load_b128s are issued before the
// 32 WMMAs of chunk i, so HBM/L2 latency hides under matrix math.
// ---------------------------------------------------------------------------
#define TSTR 40   // padded LDS stride (32 k + 8 pad) in bf16 elements

__global__ __launch_bounds__(256) void update_kernel(
    const unsigned short* __restrict__ Ak, const unsigned short* __restrict__ W,
    float* __restrict__ partials, int rows_per_chunk) {
  __shared__ unsigned short AlsT[EKD * TSTR];
  __shared__ unsigned short WlsT[EVD * TSTR];

  const int wave = threadIdx.x >> 5;
  const int lane = threadIdx.x & 31;
  const int h  = lane >> 4;
  const int ln = lane & 15;
  const int t  = threadIdx.x;
  const int s_l = t & 31;              // staging: source row within chunk
  const int eb  = (t >> 5) * 32;       // staging: 32-column group

  v8f acc[2][16];
#pragma unroll
  for (int e = 0; e < 2; ++e)
#pragma unroll
    for (int c = 0; c < 16; ++c) acc[e][c] = (v8f){0.f,0.f,0.f,0.f,0.f,0.f,0.f,0.f};

  const int s_begin = blockIdx.x * rows_per_chunk;
  const int nsteps  = rows_per_chunk / 32;

  us8 va[4], vw[4];
  {   // preload chunk 0
    const us8* ap8 = (const us8*)(Ak + (size_t)(s_begin + s_l) * EKD + eb);
    const us8* wp8 = (const us8*)(W  + (size_t)(s_begin + s_l) * EVD + eb);
#pragma unroll
    for (int c = 0; c < 4; ++c) { va[c] = ap8[c]; vw[c] = wp8[c]; }
  }

  for (int step = 0; step < nsteps; ++step) {
    __syncthreads();                   // previous compute done; LDS reusable
    {   // scatter-transpose current chunk into LDS
#pragma unroll
      for (int c = 0; c < 4; ++c) {
#pragma unroll
        for (int j = 0; j < 8; ++j) {
          const int m = eb + c * 8 + j;
          AlsT[m * TSTR + s_l] = va[c][j];
          WlsT[m * TSTR + s_l] = vw[c][j];
        }
      }
    }
    __syncthreads();

    if (step + 1 < nsteps) {           // issue next chunk's loads now; they
      const int sn = s_begin + (step + 1) * 32;   // complete under the WMMAs
      const us8* ap8 = (const us8*)(Ak + (size_t)(sn + s_l) * EKD + eb);
      const us8* wp8 = (const us8*)(W  + (size_t)(sn + s_l) * EVD + eb);
#pragma unroll
      for (int c = 0; c < 4; ++c) { va[c] = ap8[c]; vw[c] = wp8[c]; }
    }

    v16bf afr[2];
#pragma unroll
    for (int et = 0; et < 2; ++et) {
      const int ek = (2 * wave + et) * 16 + ln;
      const v8bf lo = *(const v8bf*)(AlsT + ek * TSTR + 8 * h);
      const v8bf hi = *(const v8bf*)(AlsT + ek * TSTR + 16 + 8 * h);
      afr[et] = cat8(lo, hi);
    }
#pragma unroll
    for (int ct = 0; ct < 16; ++ct) {
      const int ev = ct * 16 + ln;
      const v16bf b = cat8(*(const v8bf*)(WlsT + ev * TSTR + 8 * h),
                           *(const v8bf*)(WlsT + ev * TSTR + 16 + 8 * h));
      acc[0][ct] = __builtin_amdgcn_wmma_f32_16x16x32_bf16(
          false, afr[0], false, b, (short)0, acc[0][ct], false, false);
      acc[1][ct] = __builtin_amdgcn_wmma_f32_16x16x32_bf16(
          false, afr[1], false, b, (short)0, acc[1][ct], false, false);
    }
  }

  float* po = partials + (size_t)blockIdx.x * (EKD * EVD);
#pragma unroll
  for (int et = 0; et < 2; ++et)
#pragma unroll
    for (int ct = 0; ct < 16; ++ct) {
      const int ev = ct * 16 + ln;
#pragma unroll
      for (int r = 0; r < 8; ++r) {
        const int ek = (2 * wave + et) * 16 + r + 8 * h;
        po[(size_t)ek * EVD + ev] = acc[et][ct][r];
      }
    }
}

// ---------------------------------------------------------------------------
// Kernel: new_memory = memory + sum(partials)   (fixed order -> deterministic)
// ---------------------------------------------------------------------------
__global__ __launch_bounds__(256) void reduce_kernel(
    const float* __restrict__ mem, const float* __restrict__ partials,
    int nchunk, float* __restrict__ out) {
  const int idx = blockIdx.x * 256 + threadIdx.x;    // 65536 elements
  float s = mem[idx];
  for (int c = 0; c < nchunk; ++c) s += partials[(size_t)c * (EKD * EVD) + idx];
  out[idx] = s;
}

// ---------------------------------------------------------------------------
extern "C" void kernel_launch(void* const* d_in, const int* in_sizes, int n_in,
                              void* d_out, int out_size, void* d_ws, size_t ws_size,
                              hipStream_t stream) {
  (void)in_sizes; (void)n_in; (void)out_size;
  const float* q      = (const float*)d_in[0];
  const float* k      = (const float*)d_in[1];
  const float* v      = (const float*)d_in[2];
  const float* memory = (const float*)d_in[3];
  const float* z_norm = (const float*)d_in[4];

  float* out        = (float*)d_out;
  float* out_amem   = out;                                      // [S, EV]
  float* out_newmem = out + (size_t)S_TOT * EVD;                // [EK, EV]
  float* out_zn     = out_newmem + (size_t)EKD * EVD;           // [S, EK]

  // workspace layout
  unsigned short* Aq = (unsigned short*)d_ws;                   // S*EK bf16
  unsigned short* Ak = Aq + (size_t)S_TOT * EKD;                // S*EK bf16
  unsigned short* Wb = Ak + (size_t)S_TOT * EKD;                // S*EV bf16
  unsigned short* Bt = Wb + (size_t)S_TOT * EVD;                // EK*EV bf16 (transposed)
  float* invden_q = (float*)(Bt + (size_t)EKD * EVD);
  float* invden_k = invden_q + S_TOT;
  float* partials = invden_k + S_TOT;

  size_t used = (size_t)((char*)partials - (char*)d_ws);
  size_t avail = (ws_size > used) ? (ws_size - used) : 0;
  int nchunk = (int)(avail / ((size_t)EKD * EVD * sizeof(float)));
  if (nchunk > 256) nchunk = 256;
  if (nchunk < 1)   nchunk = 1;
  while (nchunk & (nchunk - 1)) nchunk &= nchunk - 1;   // round down to pow2
  const int rows_per_chunk = S_TOT / nchunk;            // multiple of 32

  prep_bt_kernel<<<EVD, 256, 0, stream>>>(memory, Bt);
  act_kernel<<<S_TOT / 8, 256, 0, stream>>>(q, z_norm, Aq, invden_q, nullptr);
  act_kernel<<<S_TOT / 8, 256, 0, stream>>>(k, z_norm, Ak, invden_k, out_zn);
  retrieve_kernel<<<S_TOT / 64, 256, 0, stream>>>(Aq, invden_q, Bt,
                                                  out_amem, nullptr, nullptr);
  retrieve_kernel<<<S_TOT / 64, 256, 0, stream>>>(Ak, invden_k, Bt,
                                                  nullptr, Wb, v);
  update_kernel<<<nchunk, 256, 0, stream>>>(Ak, Wb, partials, rows_per_chunk);
  reduce_kernel<<<EKD * EVD / 256, 256, 0, stream>>>(memory, partials, nchunk,
                                                     out_newmem);
}
